// TransformerTower_22041772163331
// MI455X (gfx1250) — matrix operand order repaired and checked by reading, not compiled, not run on hardware
//
#include <hip/hip_runtime.h>
#include <cstdint>
#include <cstddef>

// ============================================================================
// MI455X (gfx1250) transformer tower, wave32 + WMMA f32_16x16x32_f16.
// GEMM tiles are staged into double-buffered LDS by the Tensor Data Mover
// (tensor_load_to_lds, TENSORcnt) with hardware row padding, overlapping DMA
// with WMMA compute; transpose tiles use async global->LDS loads (ASYNCcnt).
// Norms / softmax / RoPE stay in fp32 VALU.
// ============================================================================

typedef __attribute__((ext_vector_type(16))) _Float16 v16h;
typedef __attribute__((ext_vector_type(8)))  _Float16 h8;
typedef __attribute__((ext_vector_type(8)))  float    v8f;
typedef __attribute__((ext_vector_type(4)))  unsigned u32x4;
typedef __attribute__((ext_vector_type(8)))  unsigned u32x8;

#define THREADS 256
#define BM 128
#define BN 64
#define BK 32
#define LDSA 40   // padded LDS row stride in halves (32 + 8) == TDM pad 4 DWORDs

// ---------------------------------------------------------------- reductions
__device__ __forceinline__ float block_sum(float v, float* sb) {
  const int t = threadIdx.x;
  sb[t] = v; __syncthreads();
  for (int s = THREADS / 2; s > 0; s >>= 1) {
    if (t < s) sb[t] += sb[t + s];
    __syncthreads();
  }
  float r = sb[0]; __syncthreads();
  return r;
}
__device__ __forceinline__ float block_max(float v, float* sb) {
  const int t = threadIdx.x;
  sb[t] = v; __syncthreads();
  for (int s = THREADS / 2; s > 0; s >>= 1) {
    if (t < s) sb[t] = fmaxf(sb[t], sb[t + s]);
    __syncthreads();
  }
  float r = sb[0]; __syncthreads();
  return r;
}
__device__ __forceinline__ float wave_sum(float v) {
  #pragma unroll
  for (int m = 16; m >= 1; m >>= 1) v += __shfl_xor(v, m, 32);
  return v;
}

// ------------------------------------------------ TDM 2D f16 tile -> LDS
// Loads a (rows x 32) f16 tile whose rows are row_stride halves apart into
// LDS at lds_off, inserting 4 DWORDs of padding every 16 DWORDs (so LDS row
// stride = 40 halves = LDSA). D# layout per CDNA5 ISA 8.3/8.4:
//   group0: count=1 | lds_addr | global_addr[56:0] | type=2
//   group1: wg_mask=0, data_size=1(2B), pad_enable, pad_interval=3(16 dw),
//           pad_amount=3(4 dw), tensor_dim0=row_stride, tensor_dim1=rows,
//           tile_dim0=32, tile_dim1=rows, dim0_stride=row_stride
__device__ __forceinline__ void tdm_load_tile_f16(
    const _Float16* g, unsigned lds_off, int row_stride, int rows)
{
  const unsigned long long ga = (unsigned long long)(uintptr_t)g;
  u32x4 g0;
  g0[0] = 1u;                                   // count=1, user mode
  g0[1] = lds_off;                              // lds_addr
  g0[2] = (unsigned)ga;                         // global_addr[31:0]
  g0[3] = ((unsigned)(ga >> 32) & 0x01ffffffu)  // global_addr[56:32]
          | 0x80000000u;                        // type=2 ("image")
  u32x8 g1;
  g1[0] = (1u << 16)          // data_size = 2 bytes
        | (1u << 20)          // pad_enable
        | (3u << 22)          // pad_interval: 16 DWORDs
        | (3u << 25);         // pad_amount: 4 DWORDs
  g1[1] = ((unsigned)row_stride & 0xffffu) << 16;            // dim0[15:0]
  g1[2] = ((unsigned)row_stride >> 16)
        | (((unsigned)rows & 0xffffu) << 16);                // dim0[31:16]|dim1[15:0]
  g1[3] = (((unsigned)rows >> 16) & 0xffffu) | (32u << 16);  // dim1[31:16]|tile_dim0
  g1[4] = (unsigned)rows;                                    // tile_dim1 (tile_dim2=0)
  g1[5] = (unsigned)row_stride;                              // dim0_stride[31:0]
  g1[6] = 0u;                                                // stride hi | dim1_stride lo
  g1[7] = 0u;
  asm volatile("tensor_load_to_lds %0, %1" :: "s"(g0), "s"(g1) : "memory");
}

// ------------------------------------------------ one BK-slice of WMMA work
__device__ __forceinline__ void mma_step(const _Float16* Asb,
                                         const _Float16* Bsb, v8f* acc,
                                         int w, int lm, int lk8, int lk16)
{
  // A fragment: lane holds row (w*16+lm), halves K = lk8+0..7 and lk8+16..23
  const int am = w * 16 + lm;
  h8 a0 = *(const h8*)&Asb[am * LDSA + lk8];
  h8 a1 = *(const h8*)&Asb[am * LDSA + lk8 + 16];
  v16h av = __builtin_shufflevector(a0, a1,
      0,1,2,3,4,5,6,7,8,9,10,11,12,13,14,15);
  #pragma unroll
  for (int nt = 0; nt < 4; ++nt) {
    // B fragment: lane holds col (nt*16+lm), halves K = lk8+0..7 and lk8+16..23
    const int bn = nt * 16 + lm;
    h8 b0 = *(const h8*)&Bsb[bn * LDSA + lk8];
    h8 b1 = *(const h8*)&Bsb[bn * LDSA + lk8 + 16];
    v16h bv = __builtin_shufflevector(b0, b1,
        0,1,2,3,4,5,6,7,8,9,10,11,12,13,14,15);
    acc[nt] = __builtin_amdgcn_wmma_f32_16x16x32_f16(
        false, av, false, bv, (short)0, acc[nt], false, false);
  }
}

// ---------------------------------------------------------------- WMMA GEMM
// C[M,N] (fp32) = A[M,K] (f16, lda) * Bt[N,K]^T (f16, ldb) + bias, opt ReLU.
// Batched via blockIdx.z. Block = 256 threads = 8 waves; tile 128x64.
// K must be a multiple of 64 (true for every call site). The K-loop is
// unrolled by two BK steps over double-buffered LDS: the Tensor Data Mover
// fills buffer B while the matrix pipes consume buffer A.
__global__ __launch_bounds__(THREADS) void k_gemm(
    const _Float16* __restrict__ A, const _Float16* __restrict__ Bt,
    float* __restrict__ C, const float* __restrict__ bias,
    int M, int N, int K, int lda, int ldb, int ldc,
    long strideA, long strideB, long strideC, int relu)
{
  __shared__ _Float16 As[2][BM * LDSA];
  __shared__ _Float16 Bs[2][BN * LDSA];

  const int bz = blockIdx.z;
  A  += (long)bz * strideA;
  Bt += (long)bz * strideB;
  C  += (long)bz * strideC;

  const int m0 = blockIdx.y * BM;
  const int n0 = blockIdx.x * BN;
  const int t  = threadIdx.x;
  const int w  = t >> 5;          // wave 0..7
  const int l  = t & 31;          // lane 0..31

  // WMMA fragment lane mapping (ISA 7.12.2, 16-bit A 16x32 / B 32x16)
  const int lm   = l & 15;
  const int lh   = l >> 4;        // 0 or 1
  const int lk8  = lh * 8;        // A: K-base 0 or 8
  const int lk16 = lh * 16;       // unused: B also takes K-base lk8 (0 or 8)

  v8f acc[4] = {};

  const _Float16* atile = A  + (long)m0 * lda;
  const _Float16* btile = Bt + (long)n0 * ldb;
  const unsigned as0 = (unsigned)(uintptr_t)&As[0][0];
  const unsigned as1 = (unsigned)(uintptr_t)&As[1][0];
  const unsigned bs0 = (unsigned)(uintptr_t)&Bs[0][0];
  const unsigned bs1 = (unsigned)(uintptr_t)&Bs[1][0];
  // spread L2 prefetch of upcoming K tiles across the block
  const _Float16* apre = A  + (long)(m0 + (t >> 1)) * lda + (t & 1) * 16;
  const _Float16* bpre = Bt + (long)(n0 + (t >> 2)) * ldb + (t & 3) * 8;

  if (w == 0) {   // prologue: stage first tile pair into buffer 0
    tdm_load_tile_f16(atile, as0, lda, BM);
    tdm_load_tile_f16(btile, bs0, ldb, BN);
  }

  for (int k0 = 0; k0 < K; k0 += 2 * BK) {
    // ---- consume buffer 0, TDM fills buffer 1 -------------------------
    if (w == 0) __builtin_amdgcn_s_wait_tensorcnt(0);
    __syncthreads();
    if (w == 0) {   // k0+BK < K always (K % 64 == 0)
      tdm_load_tile_f16(atile + k0 + BK, as1, lda, BM);
      tdm_load_tile_f16(btile + k0 + BK, bs1, ldb, BN);
    }
    if (k0 + 2 * BK < K) {
      __builtin_prefetch(apre + k0 + 2 * BK, 0, 1);
      __builtin_prefetch(bpre + k0 + 2 * BK, 0, 1);
    }
    mma_step(&As[0][0], &Bs[0][0], acc, w, lm, lk8, lk16);
    __syncthreads();

    // ---- consume buffer 1, TDM refills buffer 0 -----------------------
    if (w == 0) __builtin_amdgcn_s_wait_tensorcnt(0);
    __syncthreads();
    if (w == 0 && k0 + 2 * BK < K) {
      tdm_load_tile_f16(atile + k0 + 2 * BK, as0, lda, BM);
      tdm_load_tile_f16(btile + k0 + 2 * BK, bs0, ldb, BN);
    }
    mma_step(&As[1][0], &Bs[1][0], acc, w, lm, lk8, lk16);
    __syncthreads();
  }

  // epilogue: C layout per ISA (VGPR r -> M = r + lh*8, lane%16 -> N)
  #pragma unroll
  for (int nt = 0; nt < 4; ++nt) {
    const int n = n0 + nt * 16 + lm;
    const float bb = bias ? bias[n] : 0.0f;
    #pragma unroll
    for (int r = 0; r < 8; ++r) {
      const int m = m0 + w * 16 + lh * 8 + r;
      float v = acc[nt][r] + bb;
      if (relu) v = fmaxf(v, 0.0f);
      C[(long)m * ldc + n] = v;
    }
  }
  (void)M; (void)N;
}

// ------------------------------------------------- transpose + cast (f32->f16)
// dst[b][c][r] = src[b][r][c]; src: R x C f32, dst: C x R f16.
// All call-site shapes are multiples of 32, so no OOB handling is needed;
// tile loads go straight to LDS via async global->LDS (ASYNCcnt).
__global__ __launch_bounds__(THREADS) void k_tr(
    const float* __restrict__ src, _Float16* __restrict__ dst,
    int R, int Ccols, long sstride, long dstride)
{
  __shared__ float tile[32][33];
  const int b = blockIdx.z;
  src += (long)b * sstride;
  dst += (long)b * dstride;
  const int r0 = blockIdx.y * 32, c0 = blockIdx.x * 32;
  const int tx = threadIdx.x & 31, ty = threadIdx.x >> 5;
  #pragma unroll
  for (int i = 0; i < 4; ++i) {
    const int rr = ty + i * 8;
    const unsigned ldsoff =
        (unsigned)(uintptr_t)&tile[rr][tx];
    const unsigned long long ga =
        (unsigned long long)(uintptr_t)(src + (long)(r0 + rr) * Ccols + c0 + tx);
    asm volatile("global_load_async_to_lds_b32 %0, %1, off"
                 :: "v"(ldsoff), "v"(ga) : "memory");
  }
  asm volatile("s_wait_asynccnt 0x0" ::: "memory");
  __syncthreads();
  #pragma unroll
  for (int i = 0; i < 4; ++i) {
    const int c = c0 + ty + i * 8, r = r0 + tx;
    dst[(long)c * R + r] = (_Float16)tile[tx][ty + i * 8];
  }
}

// ------------------------------------------------------------------- casts
__global__ void k_cast(const float* __restrict__ src,
                       _Float16* __restrict__ dst, long n)
{
  for (long i = (long)blockIdx.x * blockDim.x + threadIdx.x; i < n;
       i += (long)gridDim.x * blockDim.x)
    dst[i] = (_Float16)src[i];
}
__global__ void k_add(float* __restrict__ dst, const float* __restrict__ src,
                      long n)
{
  for (long i = (long)blockIdx.x * blockDim.x + threadIdx.x; i < n;
       i += (long)gridDim.x * blockDim.x)
    dst[i] += src[i];
}

// ----------------------------------------------------------------- rmsnorm
__global__ __launch_bounds__(THREADS) void k_rms(
    const float* __restrict__ x, const float* __restrict__ w,
    _Float16* __restrict__ y, int d)
{
  __shared__ float sb[THREADS];
  const long row = blockIdx.x;
  const float* xr = x + row * d;
  _Float16* yr = y + row * d;
  float ss = 0.0f;
  for (int i = threadIdx.x; i < d; i += THREADS) { float v = xr[i]; ss += v * v; }
  ss = block_sum(ss, sb);
  const float inv = rsqrtf(ss / (float)d + 1.1920929e-07f);
  for (int i = threadIdx.x; i < d; i += THREADS)
    yr[i] = (_Float16)(xr[i] * inv * w[i]);
}
// resid[row] += rmsnorm(x[row], w)
__global__ __launch_bounds__(THREADS) void k_rms_add(
    const float* __restrict__ x, const float* __restrict__ w,
    float* __restrict__ resid, int d)
{
  __shared__ float sb[THREADS];
  const long row = blockIdx.x;
  const float* xr = x + row * d;
  float* rr = resid + row * d;
  float ss = 0.0f;
  for (int i = threadIdx.x; i < d; i += THREADS) { float v = xr[i]; ss += v * v; }
  ss = block_sum(ss, sb);
  const float inv = rsqrtf(ss / (float)d + 1.1920929e-07f);
  for (int i = threadIdx.x; i < d; i += THREADS)
    rr[i] += xr[i] * inv * w[i];
}

// ------------------------------------------- qkv post: LN(q,k,v) + RoPE + cast
__device__ __forceinline__ float rope_phase(int row, int d) {
  const int j = d & 63;
  const float denom = (float)j + __powf(10.0f, 1.0f + 129.0f * (float)j);
  return (float)row / denom;   // inf denom -> 0 phase, matches reference
}

__global__ __launch_bounds__(THREADS) void k_qkv_post(
    const float* __restrict__ qkv, const float* __restrict__ qn,
    const float* __restrict__ kn, const float* __restrict__ vn,
    _Float16* __restrict__ qh, _Float16* __restrict__ k16,
    float* __restrict__ vbuf, int nrows)
{
  const int row = blockIdx.x;
  const float* xr = qkv + (long)row * 1344;
  const int w = threadIdx.x >> 5;   // wave = head
  const int l = threadIdx.x & 31;

  { // q head w: layernorm(no bias) * SCALE, then RoPE
    const float* q = xr + w * 128;
    float t0 = q[l], t1 = q[l + 32], t2 = q[l + 64], t3 = q[l + 96];
    float mu  = wave_sum(t0 + t1 + t2 + t3) * (1.0f / 128.0f);
    float c0 = t0 - mu, c1 = t1 - mu, c2 = t2 - mu, c3 = t3 - mu;
    float var = wave_sum(c0*c0 + c1*c1 + c2*c2 + c3*c3) * (1.0f / 128.0f);
    float inv = rsqrtf(var + 1e-5f) * 0.125f;   // * SCALE
    float n0 = c0*inv*qn[l],    n1 = c1*inv*qn[l+32];
    float n2 = c2*inv*qn[l+64], n3 = c3*inv*qn[l+96];
    float p0 = rope_phase(row, l),      p1 = rope_phase(row, l + 32);
    float p2 = rope_phase(row, l + 64), p3 = rope_phase(row, l + 96);
    float o0 = n0*__cosf(p0) - n2*__sinf(p0);
    float o1 = n1*__cosf(p1) - n3*__sinf(p1);
    float o2 = n2*__cosf(p2) + n0*__sinf(p2);
    float o3 = n3*__cosf(p3) + n1*__sinf(p3);
    _Float16* dst = qh + ((long)w * nrows + row) * 128;
    dst[l]      = (_Float16)o0; dst[l + 32] = (_Float16)o1;
    dst[l + 64] = (_Float16)o2; dst[l + 96] = (_Float16)o3;
  }

  if (w == 0) { // k: layernorm + RoPE
    const float* kx = xr + 1024;
    float t0 = kx[l], t1 = kx[l + 32], t2 = kx[l + 64], t3 = kx[l + 96];
    float mu  = wave_sum(t0 + t1 + t2 + t3) * (1.0f / 128.0f);
    float c0 = t0 - mu, c1 = t1 - mu, c2 = t2 - mu, c3 = t3 - mu;
    float var = wave_sum(c0*c0 + c1*c1 + c2*c2 + c3*c3) * (1.0f / 128.0f);
    float inv = rsqrtf(var + 1e-5f);
    float n0 = c0*inv*kn[l],    n1 = c1*inv*kn[l+32];
    float n2 = c2*inv*kn[l+64], n3 = c3*inv*kn[l+96];
    float p0 = rope_phase(row, l),      p1 = rope_phase(row, l + 32);
    float p2 = rope_phase(row, l + 64), p3 = rope_phase(row, l + 96);
    _Float16* dst = k16 + (long)row * 128;
    dst[l]      = (_Float16)(n0*__cosf(p0) - n2*__sinf(p0));
    dst[l + 32] = (_Float16)(n1*__cosf(p1) - n3*__sinf(p1));
    dst[l + 64] = (_Float16)(n2*__cosf(p2) + n0*__sinf(p2));
    dst[l + 96] = (_Float16)(n3*__cosf(p3) + n1*__sinf(p3));
  } else if (w == 1) { // v: layernorm (192)
    const float* vx = xr + 1152;
    float tv[6]; float s = 0.0f;
    #pragma unroll
    for (int i = 0; i < 6; ++i) { tv[i] = vx[l + i * 32]; s += tv[i]; }
    float mu = wave_sum(s) * (1.0f / 192.0f);
    float vv = 0.0f;
    #pragma unroll
    for (int i = 0; i < 6; ++i) { float c = tv[i] - mu; vv += c * c; }
    float var = wave_sum(vv) * (1.0f / 192.0f);
    float inv = rsqrtf(var + 1e-5f);
    float* dst = vbuf + (long)row * 192;
    #pragma unroll
    for (int i = 0; i < 6; ++i)
      dst[l + i * 32] = (tv[i] - mu) * inv * vn[l + i * 32];
  }
}

// ----------------------------------------------------------------- softmax
__global__ __launch_bounds__(THREADS) void k_softmax(
    const float* __restrict__ S, _Float16* __restrict__ P, int cols)
{
  __shared__ float sb[THREADS];
  const long row = blockIdx.x;
  const float* x = S + row * cols;
  _Float16* y = P + row * cols;
  float mx = -3.4e38f;
  for (int i = threadIdx.x; i < cols; i += THREADS) mx = fmaxf(mx, x[i]);
  mx = block_max(mx, sb);
  float sum = 0.0f;
  for (int i = threadIdx.x; i < cols; i += THREADS) sum += __expf(x[i] - mx);
  sum = block_sum(sum, sb);
  const float inv = 1.0f / sum;
  for (int i = threadIdx.x; i < cols; i += THREADS)
    y[i] = (_Float16)(__expf(x[i] - mx) * inv);
}

// --------------------------------------------- pool(16) + exact GELU -> f16
__global__ __launch_bounds__(THREADS) void k_pool_gelu(
    const _Float16* __restrict__ x, _Float16* __restrict__ y, int D)
{
  const int orow = blockIdx.x;
  for (int c = threadIdx.x; c < D; c += THREADS) {
    float s = 0.0f;
    #pragma unroll
    for (int r = 0; r < 16; ++r)
      s += (float)x[(long)(orow * 16 + r) * D + c];
    s *= (1.0f / 16.0f);
    float g = 0.5f * s * (1.0f + erff(s * 0.70710678f));
    y[(long)orow * D + c] = (_Float16)g;
  }
}

// pw[i][j][d] (init ? = : +=) oq[i][d] + ok[j][d]   (o: 128 x 256 f32)
__global__ void k_outer(const float* __restrict__ o, float* __restrict__ pw,
                        int init)
{
  const int j = blockIdx.x, i = blockIdx.y, d = threadIdx.x;
  const float v = o[(long)i * 256 + d] + o[(long)j * 256 + 128 + d];
  const long idx = (((long)i * 128) + j) * 128 + d;
  pw[idx] = init ? v : (pw[idx] + v);
}

// ============================================================================
// host orchestration
// ============================================================================
static void launch_gemm(hipStream_t st, const _Float16* A, const _Float16* Bt,
                        float* C, const float* bias, int M, int N, int K,
                        int lda, int ldb, int ldc,
                        long sA, long sB, long sC, int batch, int relu)
{
  dim3 grid(N / BN, M / BM, batch);
  k_gemm<<<grid, THREADS, 0, st>>>(A, Bt, C, bias, M, N, K, lda, ldb, ldc,
                                   sA, sB, sC, relu);
}
static void launch_tr(hipStream_t st, const float* src, _Float16* dst,
                      int R, int Ccols, int batch, long ss, long ds)
{
  dim3 grid((Ccols + 31) / 32, (R + 31) / 32, batch);
  k_tr<<<grid, THREADS, 0, st>>>(src, dst, R, Ccols, ss, ds);
}

extern "C" void kernel_launch(void* const* d_in, const int* in_sizes, int n_in,
                              void* d_out, int out_size, void* d_ws,
                              size_t ws_size, hipStream_t stream)
{
  (void)in_sizes; (void)n_in; (void)out_size; (void)ws_size;
  const int N = 2048, DIM = 768, QKVN = 1344, HD = 1536, DFF = 1536;

  // ---- parse inputs in setup_inputs() dict order -------------------------
  int ii = 0;
  auto F = [&]() -> const float* { return (const float*)d_in[ii++]; };
  const float* x_single = F();
  struct AttnP { const float *pre, *post, *qkv, *out, *qn, *kn, *vn; };
  struct FFP   { const float *pre, *post, *w1, *b1, *w2, *b2; };
  struct PWP   { const float *s2p_pre, *s2p_outer, *a_pre, *a_qk,
                             *v_w, *v_b, *f_pre, *f_w1, *f_b1, *f_w2, *f_b2;
                 bool has; };
  AttnP ap[8]; FFP fp[8]; PWP pp[8];
  for (int L = 0; L < 8; ++L) {
    ap[L] = { F(), F(), F(), F(), F(), F(), F() };
    fp[L] = { F(), F(), F(), F(), F(), F() };
    if (L % 2 == 0) pp[L] = { F(), F(), F(), F(), F(), F(), F(), F(), F(), F(), F(), true };
    else            pp[L].has = false;
  }

  // ---- workspace bump allocator ------------------------------------------
  uintptr_t cur = (uintptr_t)d_ws;
  auto alloc = [&](size_t bytes) -> void* {
    uintptr_t r = (cur + 255) & ~(uintptr_t)255;
    cur = r + bytes;
    return (void*)r;
  };
  float*    s      = (float*)   alloc((size_t)N * DIM * 4);       // residual stream
  float*    h      = (float*)   alloc((size_t)N * HD * 4);        // temp fp32
  float*    h2     = (float*)   alloc((size_t)N * HD * 4);        // temp fp32
  float*    Sbuf   = (float*)   alloc((size_t)N * N * 4);         // scores / big fp32
  _Float16* a16    = (_Float16*)alloc((size_t)N * N * 2);         // f16 activations
  _Float16* sn16   = (_Float16*)alloc((size_t)N * DIM * 2);       // rmsnormed single
  _Float16* wt16   = (_Float16*)alloc((size_t)HD * DIM * 2);      // transposed weight
  _Float16* qh16   = (_Float16*)alloc((size_t)8 * N * 128 * 2);   // q per head
  _Float16* k16    = (_Float16*)alloc((size_t)N * 128 * 2);
  float*    vbuf   = (float*)   alloc((size_t)N * 192 * 4);
  _Float16* vT16   = (_Float16*)alloc((size_t)192 * N * 2);
  _Float16* px16   = (_Float16*)alloc((size_t)16384 * 128 * 2);   // pairwise normed
  _Float16* pvT    = (_Float16*)alloc((size_t)16384 * 128 * 2);   // pairwise v^T
  float*    pwb    = (float*)   alloc((size_t)128 * 128 * 128 * 4);
  _Float16* pool16 = (_Float16*)alloc((size_t)128 * DIM * 2);

  hipMemcpyAsync(s, x_single, (size_t)N * DIM * 4, hipMemcpyDeviceToDevice, stream);

  bool pw_live = false;
  for (int L = 0; L < 8; ++L) {
    const AttnP& A = ap[L];
    // ---------------- attention ----------------
    k_rms<<<N, THREADS, 0, stream>>>(s, A.pre, sn16, DIM);
    launch_tr(stream, A.qkv, wt16, DIM, QKVN, 1, 0, 0);
    launch_gemm(stream, sn16, wt16, h, nullptr, N, QKVN, DIM,
                DIM, DIM, QKVN, 0, 0, 0, 1, 0);
    k_qkv_post<<<N, THREADS, 0, stream>>>(h, A.qn, A.kn, A.vn,
                                          qh16, k16, vbuf, N);
    launch_tr(stream, vbuf, vT16, N, 192, 1, 0, 0);
    for (int hd = 0; hd < 8; ++hd) {    // per-head: scores, softmax, PV
      launch_gemm(stream, qh16 + (long)hd * N * 128, k16, Sbuf, nullptr,
                  N, N, 128, 128, 128, N, 0, 0, 0, 1, 0);
      k_softmax<<<N, THREADS, 0, stream>>>(Sbuf, a16, N);
      launch_gemm(stream, a16, vT16, h + (long)hd * 192, nullptr,
                  N, 192, N, N, N, HD, 0, 0, 0, 1, 0);
    }
    k_cast<<<2048, THREADS, 0, stream>>>(h, a16, (long)N * HD);
    launch_tr(stream, A.out, wt16, HD, DIM, 1, 0, 0);
    launch_gemm(stream, a16, wt16, h2, nullptr, N, DIM, HD,
                HD, HD, DIM, 0, 0, 0, 1, 0);
    k_rms_add<<<N, THREADS, 0, stream>>>(h2, A.post, s, DIM);

    // ---------------- feed-forward ----------------
    const FFP& Fp = fp[L];
    k_rms<<<N, THREADS, 0, stream>>>(s, Fp.pre, sn16, DIM);
    launch_tr(stream, Fp.w1, wt16, DIM, DFF, 1, 0, 0);
    launch_gemm(stream, sn16, wt16, h, Fp.b1, N, DFF, DIM,
                DIM, DIM, DFF, 0, 0, 0, 1, 1);
    k_cast<<<2048, THREADS, 0, stream>>>(h, a16, (long)N * DFF);
    launch_tr(stream, Fp.w2, wt16, DFF, DIM, 1, 0, 0);
    launch_gemm(stream, a16, wt16, h2, Fp.b2, N, DIM, DFF,
                DFF, DFF, DIM, 0, 0, 0, 1, 0);
    k_rms_add<<<N, THREADS, 0, stream>>>(h2, Fp.post, s, DIM);

    // ---------------- pairwise ----------------
    if (pp[L].has) {
      const PWP& P = pp[L];
      // single -> pairwise
      k_rms<<<N, THREADS, 0, stream>>>(s, P.s2p_pre, sn16, DIM);
      k_pool_gelu<<<128, THREADS, 0, stream>>>(sn16, pool16, DIM);
      launch_tr(stream, P.s2p_outer, wt16, DIM, 256, 1, 0, 0);
      launch_gemm(stream, pool16, wt16, h2, nullptr, 128, 256, DIM,
                  DIM, DIM, 256, 0, 0, 0, 1, 0);
      k_outer<<<dim3(128, 128), 128, 0, stream>>>(h2, pwb, pw_live ? 0 : 1);
      pw_live = true;
      // pairwise row attention (batched 128x128x128 WMMA GEMMs)
      k_rms<<<16384, THREADS, 0, stream>>>(pwb, P.a_pre, px16, 128);
      launch_tr(stream, P.a_qk, wt16, 128, 256, 1, 0, 0);
      launch_gemm(stream, px16, wt16, Sbuf, nullptr, 16384, 256, 128,
                  128, 128, 256, 0, 0, 0, 1, 0);          // q|k fused
      k_cast<<<2048, THREADS, 0, stream>>>(Sbuf, a16, (long)16384 * 256);
      launch_tr(stream, P.v_w, wt16, 128, 128, 1, 0, 0);
      launch_gemm(stream, px16, wt16, h, P.v_b, 16384, 128, 128,
                  128, 128, 128, 0, 0, 0, 1, 0);          // v
      launch_tr(stream, h, pvT, 128, 128, 128, 128L * 128, 128L * 128);
      launch_gemm(stream, a16, a16 + 128, Sbuf, nullptr, 128, 128, 128,
                  256, 256, 128, 128L * 256, 128L * 256, 128L * 128, 128, 0);
      k_softmax<<<16384, THREADS, 0, stream>>>(Sbuf, a16, 128);
      launch_gemm(stream, a16, pvT, h2, nullptr, 128, 128, 128,
                  128, 128, 128, 128L * 128, 128L * 128, 128L * 128, 128, 0);
      k_add<<<2048, THREADS, 0, stream>>>(pwb, h2, (long)16384 * 128);
      // pairwise feed-forward
      k_rms<<<16384, THREADS, 0, stream>>>(pwb, P.f_pre, px16, 128);
      launch_tr(stream, P.f_w1, wt16, 128, 256, 1, 0, 0);
      launch_gemm(stream, px16, wt16, Sbuf, P.f_b1, 16384, 256, 128,
                  128, 128, 256, 0, 0, 0, 1, 1);
      k_cast<<<2048, THREADS, 0, stream>>>(Sbuf, a16, (long)16384 * 256);
      launch_tr(stream, P.f_w2, wt16, 256, 128, 1, 0, 0);
      launch_gemm(stream, a16, wt16, h2, P.f_b2, 16384, 128, 256,
                  256, 256, 128, 0, 0, 0, 1, 0);
      k_add<<<2048, THREADS, 0, stream>>>(pwb, h2, (long)16384 * 128);
    }
  }

  // ---- outputs: (single, pairwise) concatenated flat ----------------------
  hipMemcpyAsync(d_out, s, (size_t)N * DIM * 4,
                 hipMemcpyDeviceToDevice, stream);
  hipMemcpyAsync((char*)d_out + (size_t)N * DIM * 4, pwb,
                 (size_t)128 * 128 * 128 * 4, hipMemcpyDeviceToDevice, stream);
}
